// PerformerAttention_19378892439765
// MI455X (gfx1250) — compile-verified, matrix operand order checked by reading, not run on hardware
//
#include <hip/hip_runtime.h>
#include <hip/hip_bf16.h>
#include <stdint.h>

// Performer attention, MI455X (gfx1250, wave32).
// All GEMMs run as v_wmma_f32_16x16x32_bf16; f32 accumulation; memory-bound design.
// Round 2: hardware bf16 converts (cast -> v_cvt), async global->LDS W-panel copy.

#define B_ 4
#define S_ 8192
#define D_ 1024
#define M_ 256
#define PANEL_STRIDE 40   // ushorts per LDS row: 32 data + 8 pad (80B) -> bank spread

typedef __bf16 bf16_t;
typedef bf16_t v16bf __attribute__((ext_vector_type(16)));
typedef float  v8f   __attribute__((ext_vector_type(8)));

union V16B { uint4 u[2]; unsigned short s[16]; v16bf v; };
union PK8  { uint4 u; unsigned short s[8]; };
union F16U { float4 f4[4]; float f[16]; };

__device__ __forceinline__ unsigned short f2bf(float f) {
  union { bf16_t h; unsigned short s; } c;
  c.h = (bf16_t)f;                      // hardware RNE convert on gfx1250
  return c.s;
}
__device__ __forceinline__ float bf2f(unsigned short s) {
  union { unsigned u; float f; } x; x.u = ((unsigned)s) << 16; return x.f;
}
__device__ __forceinline__ v8f wmma_bf16(v16bf a, v16bf b, v8f c) {
  // (neg_a, A, neg_b, B, c_mod, C, reuse_a, reuse_b)
  return __builtin_amdgcn_wmma_f32_16x16x32_bf16(false, a, false, b, (short)0, c, false, false);
}
__device__ __forceinline__ void pack8f(float4 a, float4 b, unsigned short* dst) {
  dst[0]=f2bf(a.x); dst[1]=f2bf(a.y); dst[2]=f2bf(a.z); dst[3]=f2bf(a.w);
  dst[4]=f2bf(b.x); dst[5]=f2bf(b.y); dst[6]=f2bf(b.z); dst[7]=f2bf(b.w);
}

// CDNA5 async global->LDS copy (ASYNCcnt-tracked, no VGPR round trip).
// GVS mode: mem = SADDR(sgpr64) + VADDR(u32 byte offset); LDS dst from VGPR.
__device__ __forceinline__ void async_ld_b128(unsigned lds_off, const void* sbase, unsigned goff) {
  asm volatile("global_load_async_to_lds_b128 %0, %1, %2"
               :: "v"(lds_off), "v"(goff), "s"(sbase) : "memory");
}
__device__ __forceinline__ void wait_async0() {
  asm volatile("s_wait_asynccnt 0x0" ::: "memory");
}
__device__ __forceinline__ unsigned lds_off_of(const void* p) {
  // Low 32 bits of a generic pointer to LDS == wave-relative LDS byte offset.
  return (unsigned)(size_t)p;
}

// ---------------------------------------------------------------------------
// Stage 0: W [D,M] f32  ->  Wt [M,D] bf16 ; zero k_sum accumulators.
// ---------------------------------------------------------------------------
__global__ void __launch_bounds__(256)
wt_convert_kernel(const float* __restrict__ W, unsigned short* __restrict__ Wt,
                  float* __restrict__ ksum) {
  int id = blockIdx.x * 256 + threadIdx.x;     // D_*M_ = 262144 threads
  int m = id >> 10, d = id & 1023;
  Wt[(size_t)m * D_ + d] = f2bf(W[(size_t)d * M_ + m]);
  if (id < B_ * M_) ksum[id] = 0.0f;
}

// ---------------------------------------------------------------------------
// Stage 1: X' = relu(X @ W) as bf16 WMMA GEMM.
// Workgroup: 128 S-rows x M=256.  Wave: 16 S-rows x 16 M-tiles, K-loop over D.
// W panel streamed to LDS with global_load_async_to_lds_b128.
// kmode==0: store q' row-major [b][s][m].
// kmode==1: store k'^T [b][m][s] (packed b128 per lane) + k_sum via atomics.
// ---------------------------------------------------------------------------
__global__ void __launch_bounds__(256)
features_kernel(const float* __restrict__ X, const unsigned short* __restrict__ Wt,
                unsigned short* __restrict__ outN, unsigned short* __restrict__ outT,
                float* __restrict__ ksum, int kmode) {
  __shared__ unsigned short bpanel[M_ * PANEL_STRIDE];   // 20.0 KB: W panel [m][k32]
  __shared__ float ksum_s[M_];

  const int t = threadIdx.x;
  const int wave = t >> 5, lane = t & 31;
  const int lrow = lane & 15, kSel = lane >> 4;
  const int b = blockIdx.y;
  const int sBase = blockIdx.x * 128 + wave * 16;

  if (kmode && t < M_) ksum_s[t] = 0.0f;

  v8f acc[16];
#pragma unroll
  for (int n = 0; n < 16; ++n)
#pragma unroll
    for (int j = 0; j < 8; ++j) acc[n][j] = 0.0f;

  const float* Xrow = X + ((size_t)b * S_ + (sBase + lrow)) * D_;
  const unsigned panel_lds = lds_off_of(&bpanel[t * PANEL_STRIDE]);
  const unsigned wt_row_off = (unsigned)t * (D_ * 2u);   // byte offset of row t in Wt

  for (int kb = 0; kb < D_ / 32; ++kb) {
    // Async W-panel copy: thread t owns row m=t, 32 bf16 = 4x b128, ASYNCcnt-tracked.
    {
      unsigned g = wt_row_off + (unsigned)kb * 64u;
      async_ld_b128(panel_lds +  0u, Wt, g +  0u);
      async_ld_b128(panel_lds + 16u, Wt, g + 16u);
      async_ld_b128(panel_lds + 32u, Wt, g + 32u);
      async_ld_b128(panel_lds + 48u, Wt, g + 48u);
    }
    // A fragment: lane<16 -> K 0..7 & 16..23, lane>=16 -> K 8..15 & 24..31.
    V16B a;
    {
      const float4* p0 = (const float4*)(Xrow + kb * 32 + kSel * 8);
      const float4* p1 = (const float4*)(Xrow + kb * 32 + 16 + kSel * 8);
      pack8f(p0[0], p0[1], &a.s[0]);
      pack8f(p1[0], p1[1], &a.s[8]);
    }
    if (kb + 1 < D_ / 32)
      __builtin_prefetch(Xrow + (kb + 1) * 32, 0, 1);   // global_prefetch_b8
    wait_async0();
    __syncthreads();
#pragma unroll
    for (int n = 0; n < 16; ++n) {
      V16B bm;
      const unsigned short* br = bpanel + (n * 16 + lrow) * PANEL_STRIDE + kSel * 8;
      bm.u[0] = *(const uint4*)(br);
      bm.u[1] = *(const uint4*)(br + 16);
      acc[n] = wmma_bf16(a.v, bm.v, acc[n]);
    }
    __syncthreads();
  }

  // ReLU
#pragma unroll
  for (int n = 0; n < 16; ++n)
#pragma unroll
    for (int j = 0; j < 8; ++j) acc[n][j] = fmaxf(acc[n][j], 0.0f);

  if (!kmode) {
    // q' row-major [b][s][m]; D layout: lane holds column n*16+lrow, rows i(+8).
#pragma unroll
    for (int n = 0; n < 16; ++n) {
      int col = n * 16 + lrow;
#pragma unroll
      for (int i = 0; i < 8; ++i) {
        int row = sBase + kSel * 8 + i;
        outN[((size_t)b * S_ + row) * M_ + col] = f2bf(acc[n][i]);
      }
    }
  } else {
    // k'^T [b][m][s]: lane's 8 rows are contiguous in S -> one b128 store.
#pragma unroll
    for (int n = 0; n < 16; ++n) {
      int m  = n * 16 + lrow;
      int s0 = sBase + kSel * 8;
      PK8 pk; float part = 0.0f;
#pragma unroll
      for (int i = 0; i < 8; ++i) { pk.s[i] = f2bf(acc[n][i]); part += acc[n][i]; }
      *(uint4*)(outT + ((size_t)b * M_ + m) * S_ + s0) = pk.u;
      atomicAdd(&ksum_s[m], part);
    }
    __syncthreads();
    if (t < M_) atomicAdd(&ksum[b * M_ + t], ksum_s[t]);
  }
}

// ---------------------------------------------------------------------------
// Stage 2: kv = k'^T @ v  -> kvT [b][d][m] bf16.
// Workgroup tile [64 m x 128 d]; wave tile [16 m x 64 d]; K-loop over S.
// v panel transposed to LDS [d][s] as bf16 (needs f32->bf16, so VGPR path).
// ---------------------------------------------------------------------------
__global__ void __launch_bounds__(256)
kv_kernel(const unsigned short* __restrict__ kT, const float* __restrict__ V,
          unsigned short* __restrict__ kvT) {
  __shared__ unsigned short vpanel[128 * PANEL_STRIDE];  // 10.0 KB
  const int t = threadIdx.x, wave = t >> 5, lane = t & 31;
  const int lrow = lane & 15, kSel = lane >> 4;
  const int b = blockIdx.z;
  const int mBase  = blockIdx.y * 64 + (wave >> 1) * 16;
  const int dLocal = (wave & 1) * 64;
  const int dBase  = blockIdx.x * 128 + dLocal;

  v8f acc[4];
#pragma unroll
  for (int n = 0; n < 4; ++n)
#pragma unroll
    for (int j = 0; j < 8; ++j) acc[n][j] = 0.0f;

  const int sRow   = t >> 3;          // 0..31
  const int dChunk = (t & 7) * 16;    // 0..112
  const unsigned short* kRow = kT + ((size_t)b * M_ + (mBase + lrow)) * S_;
  const float* vBase = V + (size_t)b * S_ * D_ + blockIdx.x * 128;

  for (int s = 0; s < S_; s += 32) {
    // v panel [32 s x 128 d] f32 -> LDS [d][s] bf16 (transpose during load).
    {
      F16U y;
      const float4* vp = (const float4*)(vBase + (size_t)(s + sRow) * D_ + dChunk);
      y.f4[0] = vp[0]; y.f4[1] = vp[1]; y.f4[2] = vp[2]; y.f4[3] = vp[3];
#pragma unroll
      for (int j = 0; j < 16; ++j)
        vpanel[(dChunk + j) * PANEL_STRIDE + sRow] = f2bf(y.f[j]);
    }
    // A fragment direct from k'^T (already bf16, contiguous in S).
    V16B a;
    a.u[0] = *(const uint4*)(kRow + s + kSel * 8);
    a.u[1] = *(const uint4*)(kRow + s + 16 + kSel * 8);
    if (s + 32 < S_)
      __builtin_prefetch(kRow + s + 32, 0, 1);
    __syncthreads();
#pragma unroll
    for (int n = 0; n < 4; ++n) {
      int drow = dLocal + n * 16 + lrow;
      V16B bm;
      const unsigned short* br = vpanel + drow * PANEL_STRIDE + kSel * 8;
      bm.u[0] = *(const uint4*)(br);
      bm.u[1] = *(const uint4*)(br + 16);
      acc[n] = wmma_bf16(a.v, bm.v, acc[n]);
    }
    __syncthreads();
  }

  // kvT [b][d][m]: lane's 8 rows contiguous in m -> one b128 store per tile.
#pragma unroll
  for (int n = 0; n < 4; ++n) {
    int d  = dBase + n * 16 + lrow;
    int m0 = mBase + kSel * 8;
    PK8 pk;
#pragma unroll
    for (int i = 0; i < 8; ++i) pk.s[i] = f2bf(acc[n][i]);
    *(uint4*)(kvT + ((size_t)b * D_ + d) * M_ + m0) = pk.u;
  }
}

// ---------------------------------------------------------------------------
// Stage 3: out = (q' @ kv) / (q' . k_sum + 1e-8), f32 output.
// Wave tile [16 s x 128 d], K = M = 256 (8 WMMA K-steps). Denominator computed
// per-lane alongside the A loads, combined with shfl_xor(16) + LDS broadcast.
// ---------------------------------------------------------------------------
__global__ void __launch_bounds__(256)
out_kernel(const unsigned short* __restrict__ qp, const unsigned short* __restrict__ kvT,
           const float* __restrict__ ksum, float* __restrict__ out) {
  __shared__ float den_s[8 * 16];
  const int t = threadIdx.x, wave = t >> 5, lane = t & 31;
  const int lrow = lane & 15, kSel = lane >> 4;
  const int b = blockIdx.z;
  const int sBase = blockIdx.y * 128 + wave * 16;
  const int dBase = blockIdx.x * 128;

  v8f acc[8];
#pragma unroll
  for (int n = 0; n < 8; ++n)
#pragma unroll
    for (int j = 0; j < 8; ++j) acc[n][j] = 0.0f;

  float dp = 0.0f;
  const unsigned short* qrow = qp + ((size_t)b * S_ + sBase + lrow) * M_;
  const float* ks = ksum + b * M_;

  for (int kb = 0; kb < M_ / 32; ++kb) {
    V16B a;
    a.u[0] = *(const uint4*)(qrow + kb * 32 + kSel * 8);
    a.u[1] = *(const uint4*)(qrow + kb * 32 + 16 + kSel * 8);
    // denominator partial: lane's A element j maps to k = kb*32 + (j<8 ? kSel*8+j : 16+kSel*8+j-8)
    F16U kv;
    {
      const float4* p0 = (const float4*)(ks + kb * 32 + kSel * 8);
      const float4* p1 = (const float4*)(ks + kb * 32 + 16 + kSel * 8);
      kv.f4[0] = p0[0]; kv.f4[1] = p0[1]; kv.f4[2] = p1[0]; kv.f4[3] = p1[1];
    }
#pragma unroll
    for (int j = 0; j < 16; ++j) dp += bf2f(a.s[j]) * kv.f[j];
#pragma unroll
    for (int n = 0; n < 8; ++n) {
      int d = dBase + n * 16 + lrow;
      V16B bm;
      const unsigned short* br = kvT + ((size_t)b * D_ + d) * M_ + kb * 32 + kSel * 8;
      bm.u[0] = *(const uint4*)(br);
      bm.u[1] = *(const uint4*)(br + 16);
      acc[n] = wmma_bf16(a.v, bm.v, acc[n]);
    }
  }

  // Row denominators: lanes r and r+16 each hold half the K range for row r.
  dp += __shfl_xor(dp, 16, 32);
  if (lane < 16) den_s[wave * 16 + lane] = dp;
  __syncthreads();

  float rden[8];
#pragma unroll
  for (int i = 0; i < 8; ++i)
    rden[i] = 1.0f / (den_s[wave * 16 + kSel * 8 + i] + 1e-8f);

#pragma unroll
  for (int n = 0; n < 8; ++n) {
    int d = dBase + n * 16 + lrow;
#pragma unroll
    for (int i = 0; i < 8; ++i) {
      int row = sBase + kSel * 8 + i;
      out[((size_t)b * S_ + row) * D_ + d] = acc[n][i] * rden[i];
    }
  }
}

// ---------------------------------------------------------------------------
// Workspace layout (bytes):
//   Wt   bf16 [256][1024]      :        0 ..   524288
//   q'   bf16 [4][8192][256]   :   524288 .. 17301504
//   k'^T bf16 [4][256][8192]   : 17301504 .. 34078720
//   kvT  bf16 [4][1024][256]   : 34078720 .. 36175872
//   ksum f32  [4][256]         : 36175872 .. 36179968   (~36.2 MB total)
// ---------------------------------------------------------------------------
extern "C" void kernel_launch(void* const* d_in, const int* in_sizes, int n_in,
                              void* d_out, int out_size, void* d_ws, size_t ws_size,
                              hipStream_t stream) {
  const float* q = (const float*)d_in[0];
  const float* k = (const float*)d_in[1];
  const float* v = (const float*)d_in[2];
  const float* W = (const float*)d_in[3];
  float* out = (float*)d_out;

  char* ws = (char*)d_ws;
  unsigned short* Wt  = (unsigned short*)(ws);
  unsigned short* qp  = (unsigned short*)(ws + 524288);
  unsigned short* kT  = (unsigned short*)(ws + 17301504);
  unsigned short* kvT = (unsigned short*)(ws + 34078720);
  float*          ks  = (float*)(ws + 36175872);

  wt_convert_kernel<<<dim3((D_ * M_) / 256), 256, 0, stream>>>(W, Wt, ks);
  features_kernel<<<dim3(S_ / 128, B_), 256, 0, stream>>>(q, Wt, qp, nullptr, nullptr, 0);
  features_kernel<<<dim3(S_ / 128, B_), 256, 0, stream>>>(k, Wt, nullptr, kT, ks, 1);
  kv_kernel<<<dim3(D_ / 128, M_ / 64, B_), 256, 0, stream>>>(kT, v, kvT);
  out_kernel<<<dim3(D_ / 128, S_ / 128, B_), 256, 0, stream>>>(qp, kvT, ks, out);
}
